// SEME_7791070675673
// MI455X (gfx1250) — matrix operand order repaired and checked by reading, not compiled, not run on hardware
//
#include <hip/hip_runtime.h>

// ---------------------------------------------------------------------------
// MI455X (gfx1250). Convs = implicit GEMM on v_wmma_f32_16x16x32_bf16.
// Intermediates: channel-last bf16 with 2px zero halo => conv input tiles are
// pure 2D strided copies, staged to LDS by the Tensor Data Mover
// (tensor_load_to_lds + s_wait_tensorcnt), falling back to b128 copies if the
// builtin is unavailable. ~160MB working set stays L2-resident.
// ---------------------------------------------------------------------------

#if __has_builtin(__builtin_amdgcn_tensor_load_to_lds) && __has_builtin(__builtin_amdgcn_s_wait_tensorcnt)
#define USE_TDM 1
#else
#define USE_TDM 0
#endif

typedef __attribute__((ext_vector_type(16))) __bf16 v16bf;
typedef __attribute__((ext_vector_type(8)))  float  v8f;
typedef __attribute__((ext_vector_type(4)))  unsigned v4u;
typedef __attribute__((ext_vector_type(8)))  int    v8i;
typedef __attribute__((ext_vector_type(4)))  int    v4i;

union Frag16 { v16bf v; uint4 q[2]; };

#define DEVINL __device__ __forceinline__

// Geometry: 128x128 interior, 2px halo ring (covers pad<=2), channel-last 64ch.
#define HH 128
#define WW 128
#define PM 2
#define HP (HH + 2 * PM)   // 132
#define WP (WW + 2 * PM)   // 132

DEVINL float bf2f(unsigned short h) { return __uint_as_float(((unsigned)h) << 16); }
DEVINL unsigned short f2bfbits(float f) {
  unsigned u = __float_as_uint(f);
  u += 0x7FFFu + ((u >> 16) & 1u);          // round-to-nearest-even
  return (unsigned short)(u >> 16);
}
DEVINL float sigm(float v)  { return 1.f / (1.f + __expf(-v)); }
DEVINL float lrelu(float v) { return (v >= 0.f) ? v : 0.01f * v; }
DEVINL unsigned pack2(float a, float b) {
  return (unsigned)f2bfbits(a) | ((unsigned)f2bfbits(b) << 16);
}
DEVINL void store_pack8(unsigned short* out, size_t base, v8f v) {
  uint4 st;
  st.x = pack2(v[0], v[1]); st.y = pack2(v[2], v[3]);
  st.z = pack2(v[4], v[5]); st.w = pack2(v[6], v[7]);
  *(uint4*)(out + base) = st;
}

#if USE_TDM
// 2D tile load: `rows` rows of `row_elems` bf16, global row stride
// `row_stride_elems`, into contiguous LDS at lds_addr.
DEVINL void tdm_load_2d(unsigned lds_addr, const void* gptr,
                        unsigned row_elems, unsigned rows,
                        unsigned long long row_stride_elems) {
  unsigned long long ga = (unsigned long long)gptr;
  v4u g0;
  g0.x = 1u;                                            // count=1, user D#
  g0.y = lds_addr;                                      // lds_addr
  g0.z = (unsigned)ga;                                  // global_addr[31:0]
  g0.w = (unsigned)((ga >> 32) & 0x01FFFFFFull) | 0x80000000u;  // addr[56:32]|type=2
  v8i g1;
  g1[0] = 0x10000;                                      // data_size=2B
  g1[1] = 0;                                            // tensor_dim0 lo16 (dim0=1<<30)
  g1[2] = 0x4000;                                       // tensor_dim0 hi16
  g1[3] = (int)(0x10u | (row_elems << 16));             // tensor_dim1=1<<20 | tile_dim0
  g1[4] = (int)rows;                                    // tile_dim1 (tile_dim2=0)
  g1[5] = (int)(unsigned)(row_stride_elems & 0xFFFFFFFFull);   // dim0_stride lo
  g1[6] = (int)(unsigned)((row_stride_elems >> 32) & 0xFFFFull);
  g1[7] = 0;
  v4i gz = {0, 0, 0, 0};
#if defined(__clang_major__) && __clang_major__ >= 23
  v8i gz8 = {0, 0, 0, 0, 0, 0, 0, 0};
  __builtin_amdgcn_tensor_load_to_lds(g0, g1, gz, gz, gz8, 0);
#else
  __builtin_amdgcn_tensor_load_to_lds(g0, g1, gz, gz, 0);
#endif
}
#endif

// ---------------------------------------------------------------------------
// Weight pre-swizzle into 16x32 A fragments: m = lane%16, contiguous k = 16*(lane/16) + j.
// AFrag[t][ci_chunk][co_tile][lane][16] bf16.
// ---------------------------------------------------------------------------
__global__ void prep_afrag_kernel(const float* __restrict__ w,
                                  unsigned short* __restrict__ af,
                                  int CIN, int KS) {
  int total = KS * KS * (CIN / 32) * 2048;
  int i = blockIdx.x * 256 + threadIdx.x;
  if (i >= total) return;
  int j    = i & 15;
  int lane = (i >> 4) & 31;
  int mt   = (i >> 9) & 3;
  int rest = i >> 11;
  int chunks = CIN / 32;
  int c = rest % chunks;
  int t = rest / chunks;
  int kh = t / KS, kw = t % KS;
  int m = mt * 16 + (lane & 15);
  int kk = ((lane < 16) ? 0 : 16) + j;
  int ci = c * 32 + kk;
  af[i] = f2bfbits(w[((size_t)(m * CIN + ci) * KS + kh) * KS + kw]);
}

// ---------------------------------------------------------------------------
// Generic implicit-GEMM conv (bf16 channel-last padded input) via WMMA.
// Block: 512 thr = 16 waves; block tile 64co x 4rows x 16px.
// ---------------------------------------------------------------------------
struct EpiArgs {
  const float* g;   const float* bb;  const float* kse;
  unsigned short* out0;
};

enum { EPI_BNLRELU = 0, EPI_BNLRELU_SIG = 1, EPI_BNLRELU_KSE = 2 };
enum { IN_BF16 = 0, IN_CAT = 2 };

template <int KS, int PAD, int CIN, int INMODE, int EPI>
__global__ __launch_bounds__(512) void conv_wmma(const void* __restrict__ inA,
                                                 const void* __restrict__ inB,
                                                 const unsigned short* __restrict__ afrag,
                                                 EpiArgs e) {
  constexpr int XT = 16, YT = 4;
  constexpr int IY = YT + KS - 1, IX = XT + KS - 1;
  constexpr int CHUNKS = CIN / 32;
  constexpr int REGION = IY * IX * 64;          // elems per 64-ch region
  __shared__ unsigned short tile[IY * IX * CIN];

  int tid = threadIdx.x;
  int lane = tid & 31;
  int wv = tid >> 5;
  int bid = blockIdx.x;
  int xt = bid & 7;
  int yt = (bid >> 3) & 31;
  int b  = bid >> 8;
  int x0 = xt * XT, y0 = yt * YT;

  __builtin_prefetch(afrag, 0, 0);              // global_prefetch_b8

  // ---- stage input tile (channel-last rows are contiguous) ----
  size_t goff = ((((size_t)b * HP + (y0 - PAD + PM)) * WP + (x0 - PAD + PM)) * 64);
#if USE_TDM
  if (wv == 0) {
    unsigned lds0 = (unsigned)(unsigned long long)(void*)&tile[0];
    tdm_load_2d(lds0, (const unsigned short*)inA + goff, IX * 64, IY,
                (unsigned long long)WP * 64);
    if (INMODE == IN_CAT)
      tdm_load_2d(lds0 + REGION * 2, (const unsigned short*)inB + goff,
                  IX * 64, IY, (unsigned long long)WP * 64);
    __builtin_amdgcn_s_wait_tensorcnt(0);
  }
  __syncthreads();
#else
  {
    constexpr int ROWU = IX * 64 / 8;           // uint4 per row per region
    constexpr int REGIONS = (INMODE == IN_CAT) ? 2 : 1;
    for (int u = tid; u < REGIONS * IY * ROWU; u += 512) {
      int reg = u / (IY * ROWU);
      int rem = u % (IY * ROWU);
      int iy = rem / ROWU;
      int off = (rem % ROWU) * 8;
      const unsigned short* src =
          reg ? (const unsigned short*)inB : (const unsigned short*)inA;
      size_t g = goff + (size_t)iy * WP * 64 + off;
      ((uint4*)tile)[reg * (IY * ROWU) + rem] = *(const uint4*)(src + g);
    }
  }
  __syncthreads();
#endif

  // ---- GEMM ----
  int ct = wv & 3;
  int yo = wv >> 2;
  int n = lane & 15;
  int cihalf = (lane >= 16) ? 16 : 0;
  v8f acc = {0.f, 0.f, 0.f, 0.f, 0.f, 0.f, 0.f, 0.f};

  for (int t = 0; t < KS * KS; ++t) {
    int kh = t / KS, kw = t % KS;
    int pix = (yo + kh) * IX + (n + kw);
#pragma unroll
    for (int c = 0; c < CHUNKS; ++c) {
      int base;
      if (INMODE == IN_CAT)
        base = (c >> 1) * REGION + pix * 64 + (c & 1) * 32 + cihalf;
      else
        base = pix * CIN + c * 32 + cihalf;
      Frag16 A, Bf;
      const uint4* ap = (const uint4*)(afrag +
          ((((size_t)(t * CHUNKS + c) * 4 + ct) * 32 + lane) << 4));
      A.q[0] = ap[0]; A.q[1] = ap[1];
      const uint4* bp = (const uint4*)(&tile[base]);
      Bf.q[0] = bp[0]; Bf.q[1] = bp[1];
      acc = __builtin_amdgcn_wmma_f32_16x16x32_bf16(
          false, A.v, false, Bf.v, (short)0, acc, false, false);
    }
  }

  // ---- epilogue: 8 consecutive channels per lane -> one b128 store ----
  int x = x0 + n, y = y0 + yo;
  int cob = ct * 16 + ((lane >= 16) ? 8 : 0);
  size_t obase = ((((size_t)b * HP + (y + PM)) * WP + (x + PM)) * 64) + cob;
#pragma unroll
  for (int r = 0; r < 8; ++r) {
    int co = cob + r;
    float s = e.g[co] * rsqrtf(1.f + 1e-5f);
    acc[r] = lrelu(acc[r] * s + e.bb[co]);
    if (EPI == EPI_BNLRELU_SIG) acc[r] = sigm(acc[r]);
  }
  if (EPI == EPI_BNLRELU_KSE) {
    size_t kbase = (((size_t)b * HH + y) * WW + x) * 64 + cob;
    float4 k0 = *(const float4*)&e.kse[kbase];
    float4 k1 = *(const float4*)&e.kse[kbase + 4];
    acc[0] *= 1.f + k0.x; acc[1] *= 1.f + k0.y; acc[2] *= 1.f + k0.z; acc[3] *= 1.f + k0.w;
    acc[4] *= 1.f + k1.x; acc[5] *= 1.f + k1.y; acc[6] *= 1.f + k1.z; acc[7] *= 1.f + k1.w;
  }
  store_pack8(e.out0, obase, acc);
}

// ---------------------------------------------------------------------------
// Fused three 1x1 convs over x1 (NCHW fp32): f1, f2, and the fuseadd/att path.
// One x1 tile staged (transpose+convert), three 4-chunk GEMMs reuse it.
// ---------------------------------------------------------------------------
__global__ __launch_bounds__(512) void fused1x1_kernel(
    const float* __restrict__ x1,
    const unsigned short* __restrict__ af1, const unsigned short* __restrict__ af2,
    const unsigned short* __restrict__ afv,
    const float* __restrict__ g1, const float* __restrict__ b1,
    const float* __restrict__ g2, const float* __restrict__ b2,
    const float* __restrict__ bias, const float* __restrict__ vcb,
    const float* __restrict__ kse,
    unsigned short* __restrict__ of1, unsigned short* __restrict__ of2,
    unsigned short* __restrict__ ofuse, unsigned short* __restrict__ oatt) {
  __shared__ unsigned short tile[4 * 16 * 128];
  int tid = threadIdx.x;
  int bid = blockIdx.x;
  int xt = bid & 7;
  int yt = (bid >> 3) & 31;
  int b  = bid >> 8;
  int x0 = xt * 16, y0 = yt * 4;

  for (int i = tid; i < 4 * 16 * 128; i += 512) {   // NCHW -> [iy][ix][ci] bf16
    int ci = i & 127;
    int ix = (i >> 7) & 15;
    int iy = i >> 11;
    tile[i] = f2bfbits(x1[(((size_t)b * 128 + ci) * HH + (y0 + iy)) * WW + (x0 + ix)]);
  }
  __syncthreads();

  int lane = tid & 31;
  int wv = tid >> 5;
  int ct = wv & 3;
  int yo = wv >> 2;
  int n = lane & 15;
  int cihalf = (lane >= 16) ? 16 : 0;
  int x = x0 + n, y = y0 + yo;
  int cob = ct * 16 + ((lane >= 16) ? 8 : 0);
  size_t obase = ((((size_t)b * HP + (y + PM)) * WP + (x + PM)) * 64) + cob;
  size_t kbase = (((size_t)b * HH + y) * WW + x) * 64 + cob;

#define GEMM1X1(AF, ACC)                                                      \
  _Pragma("unroll") for (int c = 0; c < 4; ++c) {                             \
    Frag16 A, Bf;                                                             \
    const uint4* ap = (const uint4*)((AF) +                                   \
        ((((size_t)c * 4 + ct) * 32 + lane) << 4));                           \
    A.q[0] = ap[0]; A.q[1] = ap[1];                                           \
    const uint4* bp = (const uint4*)(&tile[(yo * 16 + n) * 128 + c * 32 + cihalf]); \
    Bf.q[0] = bp[0]; Bf.q[1] = bp[1];                                         \
    ACC = __builtin_amdgcn_wmma_f32_16x16x32_bf16(                            \
        false, A.v, false, Bf.v, (short)0, ACC, false, false);                \
  }

  v8f a;
  // pass 1: f1 = bnlrelu(conv1(x1))
  a = (v8f){0.f,0.f,0.f,0.f,0.f,0.f,0.f,0.f};
  GEMM1X1(af1, a)
#pragma unroll
  for (int r = 0; r < 8; ++r) {
    int co = cob + r;
    a[r] = lrelu(a[r] * (g1[co] * rsqrtf(1.f + 1e-5f)) + b1[co]);
  }
  store_pack8(of1, obase, a);

  // pass 2: f2 = bnlrelu(conv2(x1))
  a = (v8f){0.f,0.f,0.f,0.f,0.f,0.f,0.f,0.f};
  GEMM1X1(af2, a)
#pragma unroll
  for (int r = 0; r < 8; ++r) {
    int co = cob + r;
    a[r] = lrelu(a[r] * (g2[co] * rsqrtf(1.f + 1e-5f)) + b2[co]);
  }
  store_pack8(of2, obase, a);

  // pass 3: fuseadd = (conv(x1)+bias + conv(va)+bias) * (1+kse); att = sigmoid
  a = (v8f){0.f,0.f,0.f,0.f,0.f,0.f,0.f,0.f};
  GEMM1X1(afv, a)
  float4 k0 = *(const float4*)&kse[kbase];
  float4 k1 = *(const float4*)&kse[kbase + 4];
  float kk[8] = {k0.x, k0.y, k0.z, k0.w, k1.x, k1.y, k1.z, k1.w};
  v8f att;
#pragma unroll
  for (int r = 0; r < 8; ++r) {
    int co = cob + r;
    a[r] = (a[r] + bias[co] + vcb[b * 64 + co]) * (1.f + kk[r]);
    att[r] = sigm(a[r]);
  }
  store_pack8(ofuse, obase, a);
  store_pack8(oatt, obase, att);
#undef GEMM1X1
}

// --------------------------- small helper kernels ---------------------------

__global__ void zero_pad_kernel(unsigned short* buf) {
  int i = blockIdx.x * 256 + threadIdx.x;      // over 8*HP*WP pixels
  if (i >= 8 * HP * WP) return;
  int xp = i % WP;
  int yp = (i / WP) % HP;
  if (yp >= PM && yp < HH + PM && xp >= PM && xp < WW + PM) return;
  uint4 z = {0, 0, 0, 0};
  uint4* p = (uint4*)buf + (size_t)i * 8;      // 64 ch = 8 x uint4
#pragma unroll
  for (int j = 0; j < 8; ++j) p[j] = z;
}

__global__ void mean_x1_kernel(const float* __restrict__ x1, float* __restrict__ va) {
  int bc = blockIdx.x;                          // b*128+ci
  const float* p = x1 + ((size_t)bc << 14);
  float s = 0.f;
  for (int i = threadIdx.x; i < 16384; i += 256) s += p[i];
  __shared__ float sm[256];
  sm[threadIdx.x] = s; __syncthreads();
  for (int o = 128; o > 0; o >>= 1) {
    if (threadIdx.x < o) sm[threadIdx.x] += sm[threadIdx.x + o];
    __syncthreads();
  }
  if (threadIdx.x == 0) va[bc] = sm[0] * (1.f / 16384.f);
}

__global__ void vconv_kernel(const float* __restrict__ va, const float* __restrict__ w,
                             const float* __restrict__ bias, float* __restrict__ vcb) {
  int i = blockIdx.x * 64 + threadIdx.x;
  int b = i >> 6, co = i & 63;
  float s = 0.f;
  for (int ci = 0; ci < 128; ++ci) s += w[co * 128 + ci] * va[b * 128 + ci];
  vcb[i] = s + bias[co];
}

// kse: bilinear 5x5 -> 128x128, stored channel-last [b][y][x][c] fp32.
__global__ void kse_kernel(const float* __restrict__ k, float* __restrict__ kse) {
  size_t i = (size_t)blockIdx.x * 256 + threadIdx.x;
  int ci = i & 63;
  int x = (i >> 6) & 127;
  int y = (i >> 13) & 127;
  int b = (int)(i >> 20);
  float fy = y * (4.f / 127.f), fx = x * (4.f / 127.f);
  int y0 = (int)floorf(fy), x0 = (int)floorf(fx);
  int y1 = (y0 + 1 < 4) ? y0 + 1 : 4;
  int x1 = (x0 + 1 < 4) ? x0 + 1 : 4;
  float wy = fy - y0, wx = fx - x0;
  const float* kb = k + ((size_t)b * 64 + ci) * 25;
  float r0 = kb[y0 * 5 + x0] * (1.f - wx) + kb[y0 * 5 + x1] * wx;
  float r1 = kb[y1 * 5 + x0] * (1.f - wx) + kb[y1 * 5 + x1] * wx;
  kse[i] = r0 * (1.f - wy) + r1 * wy;
}

// per-sample depthwise 5x5, dilations 1..3, sample0 bypass, *(1+kse). lane=ci.
__global__ void dyn_kernel(const unsigned short* __restrict__ f2p,
                           const float* __restrict__ k,
                           const float* __restrict__ kse,
                           unsigned short* __restrict__ out2p) {
  size_t i = (size_t)blockIdx.x * 256 + threadIdx.x;
  int ci = i & 63;
  int x = (i >> 6) & 127;
  int y = (i >> 13) & 127;
  int b = (int)(i >> 20);
  size_t ppix = (((size_t)b * HP + (y + PM)) * WP + (x + PM)) * 64 + ci;
  float r;
  if (b == 0) {
    r = bf2f(f2p[ppix]);
  } else {
    r = 0.f;
    const float* kb = k + ((size_t)b * 64 + ci) * 25;
    for (int d = 1; d <= 3; ++d)
      for (int t = 0; t < 25; ++t) {
        int yy = y + (t / 5 - 2) * d;
        int xx = x + (t % 5 - 2) * d;
        if (yy >= 0 && yy < HH && xx >= 0 && xx < WW)
          r += kb[t] * bf2f(f2p[(((size_t)b * HP + (yy + PM)) * WP + (xx + PM)) * 64 + ci]);
      }
  }
  r *= (1.f + kse[i]);
  out2p[ppix] = f2bfbits(r);
}

// elementwise over the FULL padded arrays (pad stays zero under mul/add).
__global__ void ew_mul3_kernel(const unsigned short* a, const unsigned short* b,
                               const unsigned short* c, unsigned short* o) {
  size_t i = (size_t)blockIdx.x * 256 + threadIdx.x;
  o[i] = f2bfbits(bf2f(a[i]) * bf2f(b[i]) * bf2f(c[i]));
}
__global__ void ew_add3_kernel(const unsigned short* a, const unsigned short* b,
                               const unsigned short* c, unsigned short* o) {
  size_t i = (size_t)blockIdx.x * 256 + threadIdx.x;
  o[i] = f2bfbits(bf2f(a[i]) + bf2f(b[i]) + bf2f(c[i]));
}

__global__ void se_mean_kernel(const unsigned short* __restrict__ att,
                               float* __restrict__ smean) {
  int bc = blockIdx.x;                          // b*64+c
  int b = bc >> 6, c = bc & 63;
  size_t base = (((size_t)b * HP + PM) * WP + PM) * 64 + c;
  float s = 0.f;
  for (int i = threadIdx.x; i < 16384; i += 256) {
    int y = i >> 7, x = i & 127;
    s += bf2f(att[base + ((size_t)y * WP + x) * 64]);
  }
  __shared__ float sm[256];
  sm[threadIdx.x] = s; __syncthreads();
  for (int o = 128; o > 0; o >>= 1) {
    if (threadIdx.x < o) sm[threadIdx.x] += sm[threadIdx.x + o];
    __syncthreads();
  }
  if (threadIdx.x == 0) smean[bc] = sm[0] * (1.f / 16384.f);
}

__global__ void se_mlp_kernel(const float* __restrict__ smean,
                              const float* __restrict__ w1,
                              const float* __restrict__ w2,
                              float* __restrict__ yse) {
  int b = blockIdx.x;
  int t = threadIdx.x;
  __shared__ float hid[4];
  if (t < 4) {
    float s = 0.f;
    for (int c = 0; c < 64; ++c) s += smean[b * 64 + c] * w1[t * 64 + c];
    hid[t] = lrelu(s);
  }
  __syncthreads();
  float s = 0.f;
  for (int h = 0; h < 4; ++h) s += hid[h] * w2[t * 4 + h];
  yse[b * 64 + t] = sigm(s);
}

// final = (att*yse) * outB * A1, written NCHW fp32.
__global__ void final_kernel(const unsigned short* __restrict__ att,
                             const float* __restrict__ yse,
                             const unsigned short* __restrict__ outB,
                             const unsigned short* __restrict__ A1,
                             float* __restrict__ out) {
  size_t i = (size_t)blockIdx.x * 256 + threadIdx.x;
  int x = i & 127;
  int y = (i >> 7) & 127;
  int co = (i >> 14) & 63;
  int b = (int)(i >> 20);
  size_t p = (((size_t)b * HP + (y + PM)) * WP + (x + PM)) * 64 + co;
  out[i] = bf2f(att[p]) * yse[b * 64 + co] * bf2f(outB[p]) * bf2f(A1[p]);
}

// --------------------------------- launch -----------------------------------

extern "C" void kernel_launch(void* const* d_in, const int* in_sizes, int n_in,
                              void* d_out, int out_size, void* d_ws, size_t ws_size,
                              hipStream_t stream) {
  (void)in_sizes; (void)n_in; (void)out_size; (void)ws_size;
  const float* x1      = (const float*)d_in[0];
  const float* k       = (const float*)d_in[1];
  const float* conv1_w = (const float*)d_in[2];
  const float* conv1_g = (const float*)d_in[3];
  const float* conv1_b = (const float*)d_in[4];
  const float* conv2_w = (const float*)d_in[5];
  const float* conv2_g = (const float*)d_in[6];
  const float* conv2_b = (const float*)d_in[7];
  const float* conv_w  = (const float*)d_in[8];
  const float* conv_bs = (const float*)d_in[9];
  const float* c0_w    = (const float*)d_in[10];
  const float* c0_g    = (const float*)d_in[11];
  const float* c0_b    = (const float*)d_in[12];
  const float* attc_w  = (const float*)d_in[13];
  const float* attc_g  = (const float*)d_in[14];
  const float* attc_b  = (const float*)d_in[15];
  const float* c05_w   = (const float*)d_in[16];
  const float* c05_g   = (const float*)d_in[17];
  const float* c05_b   = (const float*)d_in[18];
  const float* c0a_w   = (const float*)d_in[19];
  const float* c0a_g   = (const float*)d_in[20];
  const float* c0a_b   = (const float*)d_in[21];
  const float* c0b_w   = (const float*)d_in[22];
  const float* c0b_g   = (const float*)d_in[23];
  const float* c0b_b   = (const float*)d_in[24];
  const float* se_w1   = (const float*)d_in[25];
  const float* se_w2   = (const float*)d_in[26];

  char* base = (char*)d_ws;
  size_t off = 0;
  auto alloc = [&](size_t bytes) -> void* {
    void* p = base + off;
    off = (off + bytes + 255) & ~(size_t)255;
    return p;
  };
  const size_t NB  = 8ull * 64 * HH * WW;       // 8,388,608 interior elems
  const size_t PBN = 8ull * HP * WP * 64;       // 8,921,088 padded elems

  unsigned short* buf[8];
  for (int i = 0; i < 8; ++i) buf[i] = (unsigned short*)alloc(PBN * 2);
  float* kse = (float*)alloc(NB * 4);
  unsigned short* af_c0   = (unsigned short*)alloc((size_t)9 * 2 * 2048 * 2);
  unsigned short* af_attc = (unsigned short*)alloc((size_t)9 * 2 * 2048 * 2);
  unsigned short* af_c05  = (unsigned short*)alloc((size_t)25 * 2 * 2048 * 2);
  unsigned short* af_c0b  = (unsigned short*)alloc((size_t)9 * 2 * 2048 * 2);
  unsigned short* af_c0a  = (unsigned short*)alloc((size_t)1 * 4 * 2048 * 2);
  unsigned short* af_c1   = (unsigned short*)alloc((size_t)1 * 4 * 2048 * 2);
  unsigned short* af_c2   = (unsigned short*)alloc((size_t)1 * 4 * 2048 * 2);
  unsigned short* af_cv   = (unsigned short*)alloc((size_t)1 * 4 * 2048 * 2);
  float* va    = (float*)alloc(8 * 128 * 4);
  float* vcb   = (float*)alloc(8 * 64 * 4);
  float* smean = (float*)alloc(8 * 64 * 4);
  float* yse   = (float*)alloc(8 * 64 * 4);

  auto prep = [&](const float* w, unsigned short* af, int CIN, int KS) {
    int total = KS * KS * (CIN / 32) * 2048;
    prep_afrag_kernel<<<(total + 255) / 256, 256, 0, stream>>>(w, af, CIN, KS);
  };
  prep(c0_w,    af_c0,   64, 3);
  prep(attc_w,  af_attc, 64, 3);
  prep(c05_w,   af_c05,  64, 5);
  prep(c0b_w,   af_c0b,  64, 3);
  prep(c0a_w,   af_c0a, 128, 1);
  prep(conv1_w, af_c1,  128, 1);
  prep(conv2_w, af_c2,  128, 1);
  prep(conv_w,  af_cv,  128, 1);

  const int gPad = (8 * HP * WP + 255) / 256;
  for (int i = 0; i < 8; ++i) zero_pad_kernel<<<gPad, 256, 0, stream>>>(buf[i]);

  mean_x1_kernel<<<1024, 256, 0, stream>>>(x1, va);
  vconv_kernel<<<8, 64, 0, stream>>>(va, conv_w, conv_bs, vcb);
  const int gEW  = (int)(NB / 256);
  const int gEWP = (int)((PBN + 255) / 256);
  kse_kernel<<<gEW, 256, 0, stream>>>(k, kse);

  dim3 gc(2048), bc(512);
  // Buffer plan: b0:f1->out2conv  b1:f2->outA  b2:fuseadd->out2->finout2->out2f
  //              b3:att  b4:eadd->A2->eout2->outB  b5:A1  b6:out1  b7:fuseout2
  fused1x1_kernel<<<gc, bc, 0, stream>>>(x1, af_c1, af_c2, af_cv,
      conv1_g, conv1_b, conv2_g, conv2_b, conv_bs, vcb, kse,
      buf[0], buf[1], buf[2], buf[3]);

  EpiArgs e;
  e = {}; e.g = c0_g; e.bb = c0_b; e.out0 = buf[4];        // eadd = c0(fuseadd)
  conv_wmma<3, 1, 64, IN_BF16, EPI_BNLRELU><<<gc, bc, 0, stream>>>(buf[2], nullptr, af_c0, e);

  e = {}; e.g = attc_g; e.bb = attc_b; e.out0 = buf[5];    // A1 = sig(attc(eadd))
  conv_wmma<3, 1, 64, IN_BF16, EPI_BNLRELU_SIG><<<gc, bc, 0, stream>>>(buf[4], nullptr, af_attc, e);

  e = {}; e.g = c0_g; e.bb = c0_b; e.kse = kse; e.out0 = buf[6];   // out1
  conv_wmma<3, 1, 64, IN_BF16, EPI_BNLRELU_KSE><<<gc, bc, 0, stream>>>(buf[0], nullptr, af_c0, e);

  e = {}; e.g = attc_g; e.bb = attc_b; e.out0 = buf[4];    // A2 = sig(attc(out1))
  conv_wmma<3, 1, 64, IN_BF16, EPI_BNLRELU_SIG><<<gc, bc, 0, stream>>>(buf[6], nullptr, af_attc, e);

  dyn_kernel<<<gEW, 256, 0, stream>>>(buf[1], k, kse, buf[2]);     // out2

  e = {}; e.g = c05_g; e.bb = c05_b; e.out0 = buf[0];      // out2conv = c05(out2)
  conv_wmma<5, 2, 64, IN_BF16, EPI_BNLRELU><<<gc, bc, 0, stream>>>(buf[2], nullptr, af_c05, e);

  ew_mul3_kernel<<<gEWP, 256, 0, stream>>>(buf[5], buf[4], buf[0], buf[7]); // fuseout2

  e = {}; e.g = c05_g; e.bb = c05_b; e.out0 = buf[2];      // finout2
  conv_wmma<5, 2, 64, IN_BF16, EPI_BNLRELU><<<gc, bc, 0, stream>>>(buf[7], nullptr, af_c05, e);

  ew_add3_kernel<<<gEWP, 256, 0, stream>>>(buf[2], buf[0], buf[1], buf[4]); // eout2

  e = {}; e.g = c05_g; e.bb = c05_b; e.out0 = buf[2];      // out2 final
  conv_wmma<5, 2, 64, IN_BF16, EPI_BNLRELU><<<gc, bc, 0, stream>>>(buf[4], nullptr, af_c05, e);

  e = {}; e.g = c0a_g; e.bb = c0a_b; e.out0 = buf[1];      // outA = conv0a(cat)
  conv_wmma<1, 0, 128, IN_CAT, EPI_BNLRELU><<<gc, bc, 0, stream>>>(buf[6], buf[2], af_c0a, e);

  e = {}; e.g = c0b_g; e.bb = c0b_b; e.out0 = buf[4];      // outB = conv0b(outA)
  conv_wmma<3, 1, 64, IN_BF16, EPI_BNLRELU><<<gc, bc, 0, stream>>>(buf[1], nullptr, af_c0b, e);

  se_mean_kernel<<<512, 256, 0, stream>>>(buf[3], smean);
  se_mlp_kernel<<<8, 64, 0, stream>>>(smean, se_w1, se_w2, yse);

  final_kernel<<<gEW, 256, 0, stream>>>(buf[3], yse, buf[4], buf[5], (float*)d_out);
}
